// GATOnlyBaseline_17781164606103
// MI455X (gfx1250) — compile-verified, matrix operand order checked by reading, not run on hardware
//
#include <hip/hip_runtime.h>
#include <hip/hip_bf16.h>

typedef __attribute__((ext_vector_type(16))) _Float16 v16h;
typedef __attribute__((ext_vector_type(4)))  _Float16 v4h;
typedef __attribute__((ext_vector_type(8)))  float    v8f;
typedef __attribute__((ext_vector_type(4)))  float    v4f;

#define LDSS 40  // half-elements per LDS row (32 data + 8 pad)

// ---------------------------------------------------------------------------
// WMMA GEMM: C[M,N] = act(A[M,K] @ B[K,N] + bias), all f32 in global,
// converted to f16 in LDS, accumulated in f32 via v_wmma_f32_16x16x32_f16.
// Block = 256 threads = 8 waves, computes 64 x BN tile. BN in {32, 64}.
// ---------------------------------------------------------------------------
template<int BN, int ACT>
__global__ __launch_bounds__(256) void k_gemm_wmma(
    const float* __restrict__ A, const float* __restrict__ B,
    const float* __restrict__ bias, float* __restrict__ C,
    int M, int N, int K)
{
  __shared__ _Float16 lA[64 * LDSS];
  __shared__ _Float16 lB[BN * LDSS];
  constexpr int TPW = BN / 32;          // 16x16 C tiles per wave (1 or 2)
  const int tid  = threadIdx.x;
  const int lane = tid & 31;
  const int wave = tid >> 5;
  const int wr   = wave >> 1;           // 0..3 -> M sub-tile
  const int wc   = wave & 1;            // 0..1 -> N half
  const int m0   = blockIdx.x * 64;
  const int n0   = blockIdx.y * BN;

  v8f acc[TPW];
#pragma unroll
  for (int t = 0; t < TPW; ++t) acc[t] = v8f{0,0,0,0,0,0,0,0};

  const int fm = lane & 15;             // row (A) / col (B) within tile
  const int kb = (lane >> 4) * 8;       // K base per ISA 16-bit layout

  for (int k0 = 0; k0 < K; k0 += 32) {
    __syncthreads();
    // ---- stage A tile (64 x 32), f32 -> f16 ----
    {
      const int r  = tid >> 2;
      const int c  = (tid & 3) * 8;
      const int gm = m0 + r;
      float v[8];
      if (gm < M) {
        const float* p = A + (size_t)gm * K + k0 + c;
        v4f x0 = *(const v4f*)p;
        v4f x1 = *(const v4f*)(p + 4);
#pragma unroll
        for (int j = 0; j < 4; ++j) { v[j] = x0[j]; v[4 + j] = x1[j]; }
      } else {
#pragma unroll
        for (int j = 0; j < 8; ++j) v[j] = 0.f;
      }
      _Float16* d = lA + r * LDSS + c;
#pragma unroll
      for (int j = 0; j < 8; ++j) d[j] = (_Float16)v[j];
    }
    // ---- stage B tile (32 x BN), transposed to lB[n][k] ----
    {
      constexpr int PER = BN / 8;
      const int kk  = tid >> 3;
      const int nst = (tid & 7) * PER;
      const float* p = B + (size_t)(k0 + kk) * N + n0 + nst;
#pragma unroll
      for (int j0 = 0; j0 < PER; j0 += 4) {
        v4f x = *(const v4f*)(p + j0);
#pragma unroll
        for (int jj = 0; jj < 4; ++jj)
          lB[(nst + j0 + jj) * LDSS + kk] = (_Float16)x[jj];
      }
    }
    __syncthreads();
    // ---- build fragments (ISA 16-bit 16x32 layout) + WMMA ----
    v16h af;
    {
      const _Float16* pa = lA + (wr * 16 + fm) * LDSS + kb;
      v4h a0 = *(const v4h*)(pa);
      v4h a1 = *(const v4h*)(pa + 4);
      v4h a2 = *(const v4h*)(pa + 16);
      v4h a3 = *(const v4h*)(pa + 20);
#pragma unroll
      for (int j = 0; j < 4; ++j) {
        af[j] = a0[j]; af[4 + j] = a1[j]; af[8 + j] = a2[j]; af[12 + j] = a3[j];
      }
    }
#pragma unroll
    for (int t = 0; t < TPW; ++t) {
      const _Float16* pb = lB + (wc * (BN / 2) + t * 16 + fm) * LDSS + kb;
      v4h b0 = *(const v4h*)(pb);
      v4h b1 = *(const v4h*)(pb + 4);
      v4h b2 = *(const v4h*)(pb + 16);
      v4h b3 = *(const v4h*)(pb + 20);
      v16h bf;
#pragma unroll
      for (int j = 0; j < 4; ++j) {
        bf[j] = b0[j]; bf[4 + j] = b1[j]; bf[8 + j] = b2[j]; bf[12 + j] = b3[j];
      }
      acc[t] = __builtin_amdgcn_wmma_f32_16x16x32_f16(
          false, af, false, bf, (short)0, acc[t], false, false);
    }
  }
  // ---- epilogue: bias + optional ELU, per 16x16 f32 D layout ----
  const int row_lo = m0 + wr * 16 + (lane >> 4) * 8;
#pragma unroll
  for (int t = 0; t < TPW; ++t) {
    const int col = n0 + wc * (BN / 2) + t * 16 + fm;
    const float bia = bias ? bias[col] : 0.f;
#pragma unroll
    for (int r = 0; r < 8; ++r) {
      const int row = row_lo + r;
      if (row < M) {
        float v = acc[t][r] + bia;
        if (ACT == 1) v = v > 0.f ? v : (__expf(v) - 1.f);
        C[(size_t)row * N + col] = v;
      }
    }
  }
}

// ---------------------------------------------------------------------------
// Small helper kernels
// ---------------------------------------------------------------------------
__global__ void k_fill(float* __restrict__ p, float v, size_t n) {
  size_t i = (size_t)blockIdx.x * blockDim.x + threadIdx.x;
  if (i < n) p[i] = v;
}

// V[k*Hh+h] = sum_c W[k, h*C+c] * a[h*C+c]   (attention precontraction)
__global__ void k_compute_V(const float* __restrict__ W, const float* __restrict__ a,
                            float* __restrict__ V, int Krows, int Hh, int C) {
  int i = blockIdx.x * blockDim.x + threadIdx.x;
  if (i >= Krows * Hh) return;
  int k = i / Hh, h = i % Hh;
  float s = 0.f;
  for (int c = 0; c < C; ++c) s += W[(size_t)k * (Hh * C) + h * C + c] * a[h * C + c];
  V[i] = s;
}

// out[n*Hh+h] = sum_k X[n,k] * V[k*Hh+h]
__global__ void k_node_dot(const float* __restrict__ X, const float* __restrict__ V,
                           float* __restrict__ out, int N, int Kdim, int Hh) {
  int i = blockIdx.x * blockDim.x + threadIdx.x;
  if (i >= N * Hh) return;
  int n = i / Hh, h = i % Hh;
  float s = 0.f;
  const float* x = X + (size_t)n * Kdim;
  for (int k = 0; k < Kdim; ++k) s += x[k] * V[k * Hh + h];
  out[i] = s;
}

__device__ inline void atomicMaxF(float* addr, float v) {
  if (v >= 0.f) atomicMax((int*)addr, __float_as_int(v));
  else          atomicMin((unsigned int*)addr, __float_as_uint(v));
}

__global__ void k_edge_logits(const int* __restrict__ src, const int* __restrict__ dst,
                              const float* __restrict__ als, const float* __restrict__ ald,
                              float* __restrict__ ebuf, float* __restrict__ emax,
                              int E_, int Hh) {
  int i = blockIdx.x * blockDim.x + threadIdx.x;
  if (i >= E_ * Hh) return;
  int e = i / Hh, h = i % Hh;
  int s = src[e], d = dst[e];
  float x = als[s * Hh + h] + ald[d * Hh + h];
  x = x > 0.f ? x : 0.2f * x;               // leaky_relu, slope 0.2
  ebuf[i] = x;
  atomicMaxF(&emax[d * Hh + h], x);
}

__global__ void k_edge_exp(const int* __restrict__ dst, float* __restrict__ ebuf,
                           const float* __restrict__ emax, float* __restrict__ den,
                           int E_, int Hh) {
  int i = blockIdx.x * blockDim.x + threadIdx.x;
  if (i >= E_ * Hh) return;
  int e = i / Hh, h = i % Hh;
  int d = dst[e];
  float ex = __expf(ebuf[i] - emax[d * Hh + h]);
  ebuf[i] = ex;
  atomicAdd(&den[d * Hh + h], ex);
}

__global__ void k_edge_agg(const int* __restrict__ src, const int* __restrict__ dst,
                           const float* __restrict__ ebuf, const float* __restrict__ den,
                           const float* __restrict__ hs, float* __restrict__ out,
                           int E_, int Hh, int C) {
  int i = blockIdx.x * blockDim.x + threadIdx.x;
  if (i >= E_ * Hh) return;
  int e = i / Hh, h = i % Hh;
  int s = src[e], d = dst[e];
  float alpha = ebuf[i] / (den[d * Hh + h] + 1e-16f);
  const float* ps = hs + (size_t)s * Hh * C + h * C;
  float* pd = out + (size_t)d * Hh * C + h * C;
  for (int c = 0; c < C; ++c) atomicAdd(&pd[c], alpha * ps[c]);
}

__global__ void k_bias_elu(float* __restrict__ X, const float* __restrict__ b,
                           int N, int F) {
  size_t i = (size_t)blockIdx.x * blockDim.x + threadIdx.x;
  if (i >= (size_t)N * F) return;
  int f = (int)(i % F);
  float v = X[i] + b[f];
  X[i] = v > 0.f ? v : (__expf(v) - 1.f);
}

__global__ void k_classifier(const float* __restrict__ t2, const float* __restrict__ Wc,
                             const float* __restrict__ bc, float* __restrict__ out,
                             int N, int F) {
  int n = blockIdx.x * blockDim.x + threadIdx.x;
  if (n >= N) return;
  float s = bc[0];
  const float* p = t2 + (size_t)n * F;
  for (int f = 0; f < F; ++f) s += p[f] * Wc[f];
  out[n] = s;
}

// ---------------------------------------------------------------------------
extern "C" void kernel_launch(void* const* d_in, const int* in_sizes, int n_in,
                              void* d_out, int out_size, void* d_ws, size_t ws_size,
                              hipStream_t stream) {
  constexpr int N_TX = 100000, N_USER = 50000, F_TX = 128, F_USER = 64;
  constexpr int HID = 64, H = 4, OUT = 32, E = 500000;
  (void)in_sizes; (void)n_in; (void)out_size; (void)ws_size;

  const float* x_tx    = (const float*)d_in[0];
  const float* x_user  = (const float*)d_in[1];
  const int*   ei_u2t  = (const int*)d_in[2];
  const int*   ei_t2u  = (const int*)d_in[3];
  const float* Wp_tx   = (const float*)d_in[4];
  const float* bp_tx   = (const float*)d_in[5];
  const float* Wp_us   = (const float*)d_in[6];
  const float* bp_us   = (const float*)d_in[7];
  const float* W1_u2t  = (const float*)d_in[8];
  const float* as1_u2t = (const float*)d_in[9];
  const float* ad1_u2t = (const float*)d_in[10];
  const float* b1_u2t  = (const float*)d_in[11];
  const float* W2_u2t  = (const float*)d_in[12];
  const float* as2_u2t = (const float*)d_in[13];
  const float* ad2_u2t = (const float*)d_in[14];
  const float* b2_u2t  = (const float*)d_in[15];
  const float* W1_t2u  = (const float*)d_in[16];
  const float* as1_t2u = (const float*)d_in[17];
  const float* ad1_t2u = (const float*)d_in[18];
  const float* b1_t2u  = (const float*)d_in[19];
  // d_in[20..23] = W2_t2u etc: unused (only transaction output feeds classifier)
  const float* Wc      = (const float*)d_in[24];
  const float* bc      = (const float*)d_in[25];

  const int* src_u = ei_u2t;      const int* dst_t = ei_u2t + E;
  const int* src_t = ei_t2u;      const int* dst_u = ei_t2u + E;

  // workspace carving (lifetimes are sequential where buffers are shared)
  char* w = (char*)d_ws; size_t off = 0;
  auto alloc = [&](size_t nf) -> float* {
    float* p = (float*)(w + off);
    off += ((nf * sizeof(float) + 255) & ~(size_t)255);
    return p;
  };
  float* h_tx = alloc((size_t)N_TX * HID);
  float* h_us = alloc((size_t)N_USER * HID);
  float* HS   = alloc((size_t)N_TX * H * HID);   // src features, reused per conv
  float* t1   = alloc((size_t)N_TX * H * HID);
  float* u1   = alloc((size_t)N_USER * H * HID);
  float* hs2  = alloc((size_t)N_USER * OUT);
  float* t2   = alloc((size_t)N_TX * OUT);
  float* ebuf = alloc((size_t)E * H);
  float* alsA = alloc((size_t)N_TX * H);
  float* aldA = alloc((size_t)N_TX * H);
  float* emax = alloc((size_t)N_TX * H);
  float* den  = alloc((size_t)N_TX * H);
  float* Vs1u = alloc(HID * H);
  float* Vd1u = alloc(HID * H);
  float* Vs1t = alloc(HID * H);
  float* Vd1t = alloc(HID * H);
  float* V2d  = alloc(H * HID);

  dim3 blk(256);
  auto g1 = [](long n) { return dim3((unsigned)((n + 255) / 256)); };
  const float NEG_INF = -__builtin_huge_valf();

  // projections (fused bias + ELU)
  k_gemm_wmma<64,1><<<dim3((N_TX+63)/64, HID/64), blk, 0, stream>>>(x_tx, Wp_tx, bp_tx, h_tx, N_TX, HID, F_TX);
  k_gemm_wmma<64,1><<<dim3((N_USER+63)/64, HID/64), blk, 0, stream>>>(x_user, Wp_us, bp_us, h_us, N_USER, HID, F_USER);

  // attention precontractions (eliminates destination-side full GEMMs)
  k_compute_V<<<g1(HID*H), blk, 0, stream>>>(W1_u2t, as1_u2t, Vs1u, HID, H, HID);
  k_compute_V<<<g1(HID*H), blk, 0, stream>>>(W1_u2t, ad1_u2t, Vd1u, HID, H, HID);
  k_compute_V<<<g1(HID*H), blk, 0, stream>>>(W1_t2u, as1_t2u, Vs1t, HID, H, HID);
  k_compute_V<<<g1(HID*H), blk, 0, stream>>>(W1_t2u, ad1_t2u, Vd1t, HID, H, HID);
  k_compute_V<<<g1(H*HID), blk, 0, stream>>>(W2_u2t, ad2_u2t, V2d, H*HID, 1, OUT);

  // ---- conv1: user -> transaction (heads=4, concat) ----
  k_gemm_wmma<64,0><<<dim3((N_USER+63)/64, (H*HID)/64), blk, 0, stream>>>(h_us, W1_u2t, nullptr, HS, N_USER, H*HID, HID);
  k_node_dot<<<g1((long)N_USER*H), blk, 0, stream>>>(h_us, Vs1u, alsA, N_USER, HID, H);
  k_node_dot<<<g1((long)N_TX*H),   blk, 0, stream>>>(h_tx, Vd1u, aldA, N_TX, HID, H);
  k_fill<<<g1((long)N_TX*H), blk, 0, stream>>>(emax, NEG_INF, (size_t)N_TX*H);
  k_fill<<<g1((long)N_TX*H), blk, 0, stream>>>(den, 0.f, (size_t)N_TX*H);
  k_fill<<<g1((long)N_TX*H*HID), blk, 0, stream>>>(t1, 0.f, (size_t)N_TX*H*HID);
  k_edge_logits<<<g1((long)E*H), blk, 0, stream>>>(src_u, dst_t, alsA, aldA, ebuf, emax, E, H);
  k_edge_exp<<<g1((long)E*H), blk, 0, stream>>>(dst_t, ebuf, emax, den, E, H);
  k_edge_agg<<<g1((long)E*H), blk, 0, stream>>>(src_u, dst_t, ebuf, den, HS, t1, E, H, HID);
  k_bias_elu<<<g1((long)N_TX*H*HID), blk, 0, stream>>>(t1, b1_u2t, N_TX, H*HID);

  // ---- conv1: transaction -> user ----
  k_gemm_wmma<64,0><<<dim3((N_TX+63)/64, (H*HID)/64), blk, 0, stream>>>(h_tx, W1_t2u, nullptr, HS, N_TX, H*HID, HID);
  k_node_dot<<<g1((long)N_TX*H),   blk, 0, stream>>>(h_tx, Vs1t, alsA, N_TX, HID, H);
  k_node_dot<<<g1((long)N_USER*H), blk, 0, stream>>>(h_us, Vd1t, aldA, N_USER, HID, H);
  k_fill<<<g1((long)N_USER*H), blk, 0, stream>>>(emax, NEG_INF, (size_t)N_USER*H);
  k_fill<<<g1((long)N_USER*H), blk, 0, stream>>>(den, 0.f, (size_t)N_USER*H);
  k_fill<<<g1((long)N_USER*H*HID), blk, 0, stream>>>(u1, 0.f, (size_t)N_USER*H*HID);
  k_edge_logits<<<g1((long)E*H), blk, 0, stream>>>(src_t, dst_u, alsA, aldA, ebuf, emax, E, H);
  k_edge_exp<<<g1((long)E*H), blk, 0, stream>>>(dst_u, ebuf, emax, den, E, H);
  k_edge_agg<<<g1((long)E*H), blk, 0, stream>>>(src_t, dst_u, ebuf, den, HS, u1, E, H, HID);
  k_bias_elu<<<g1((long)N_USER*H*HID), blk, 0, stream>>>(u1, b1_t2u, N_USER, H*HID);

  // ---- conv2: user -> transaction (heads=1, mean==identity) ----
  k_gemm_wmma<32,0><<<dim3((N_USER+63)/64, OUT/32), blk, 0, stream>>>(u1, W2_u2t, nullptr, hs2, N_USER, OUT, H*HID);
  k_node_dot<<<g1((long)N_USER), blk, 0, stream>>>(hs2, as2_u2t, alsA, N_USER, OUT, 1);
  k_node_dot<<<g1((long)N_TX),   blk, 0, stream>>>(t1, V2d, aldA, N_TX, H*HID, 1);
  k_fill<<<g1((long)N_TX), blk, 0, stream>>>(emax, NEG_INF, (size_t)N_TX);
  k_fill<<<g1((long)N_TX), blk, 0, stream>>>(den, 0.f, (size_t)N_TX);
  k_fill<<<g1((long)N_TX*OUT), blk, 0, stream>>>(t2, 0.f, (size_t)N_TX*OUT);
  k_edge_logits<<<g1((long)E), blk, 0, stream>>>(src_u, dst_t, alsA, aldA, ebuf, emax, E, 1);
  k_edge_exp<<<g1((long)E), blk, 0, stream>>>(dst_t, ebuf, emax, den, E, 1);
  k_edge_agg<<<g1((long)E), blk, 0, stream>>>(src_u, dst_t, ebuf, den, hs2, t2, E, 1, OUT);
  k_bias_elu<<<g1((long)N_TX*OUT), blk, 0, stream>>>(t2, b2_u2t, N_TX, OUT);

  // classifier
  k_classifier<<<g1((long)N_TX), blk, 0, stream>>>(t2, Wc, bc, (float*)d_out, N_TX, OUT);
}
